// Cotta_Adapter_23596550324687
// MI455X (gfx1250) — matrix operand order repaired and checked by reading, not compiled, not run on hardware
//
#include <hip/hip_runtime.h>
#include <hip/hip_bf16.h>
#include <stdint.h>

typedef __attribute__((ext_vector_type(16))) __bf16 bf16x16;
typedef __attribute__((ext_vector_type(8)))  float  f32x8;

#define N_TOK 32768
#define DIM   1024
#define HID   192
#define NEXP  4
#define ADPT_SCALE 0.8f

// ---- workspace layout (bytes) ----
#define OFF_W2    0u
#define OFF_PART  (OFF_W2   + (unsigned)N_TOK * 4u * 4u)        // 524288
#define OFF_KD    (OFF_PART + 4096u * 4u * 4u)                  // 589824
#define OFF_DWH   (OFF_KD   + 4096u)                            // 593920
#define OFF_DWL   (OFF_DWH  + (unsigned)NEXP * HID * DIM * 2u)
#define OFF_UWH   (OFF_DWL  + (unsigned)NEXP * HID * DIM * 2u)
#define OFF_UWL   (OFF_UWH  + (unsigned)NEXP * DIM * HID * 2u)

// ---- helpers ----
__device__ __forceinline__ unsigned short f2bf(float f) {
  unsigned u = __float_as_uint(f);
  unsigned r = u + 0x7FFFu + ((u >> 16) & 1u);   // RNE
  return (unsigned short)(r >> 16);
}
__device__ __forceinline__ float bf2f(unsigned short h) {
  return __uint_as_float(((unsigned)h) << 16);
}
__device__ __forceinline__ void splitbf(float v, unsigned short& h, unsigned short& l) {
  h = f2bf(v);
  l = f2bf(v - bf2f(h));
}
// order-preserving float -> uint key
__device__ __forceinline__ unsigned f2key(float f) {
  unsigned u = __float_as_uint(f);
  return (u & 0x80000000u) ? ~u : (u | 0x80000000u);
}
__device__ __forceinline__ float key2f(unsigned k) {
  unsigned u = (k & 0x80000000u) ? (k ^ 0x80000000u) : ~k;
  return __uint_as_float(u);
}
__device__ __forceinline__ int wred_i(int v) {
  #pragma unroll
  for (int o = 16; o > 0; o >>= 1) v += __shfl_xor(v, o, 32);
  return v;
}
__device__ __forceinline__ float wred_f(float v) {
  #pragma unroll
  for (int o = 16; o > 0; o >>= 1) v += __shfl_xor(v, o, 32);
  return v;
}
// value of s[rank] (0-indexed ascending) over wave-distributed keys, via 32-step bisection
template <int NV>
__device__ __forceinline__ unsigned rank_select(const unsigned (&keys)[NV], int rank) {
  unsigned lo = 0u, hi = 0xFFFFFFFFu;
  for (int it = 0; it < 32; ++it) {
    unsigned mid = lo + ((hi - lo) >> 1);
    int c = 0;
    #pragma unroll
    for (int j = 0; j < NV; ++j) c += (keys[j] <= mid) ? 1 : 0;
    c = wred_i(c);
    if (c > rank) hi = mid; else lo = mid + 1u;
  }
  return lo;
}
__device__ __forceinline__ void top2softmax(const float* l, float* w) {
  float m1 = fmaxf(fmaxf(l[0], l[1]), fmaxf(l[2], l[3]));
  int c = 0;
  #pragma unroll
  for (int e = 0; e < 4; ++e) c += (l[e] == m1) ? 1 : 0;
  float kth;
  if (c >= 2) { kth = m1; }
  else {
    kth = -3.4e38f;
    #pragma unroll
    for (int e = 0; e < 4; ++e) if (l[e] < m1) kth = fmaxf(kth, l[e]);
  }
  float s = 0.f;
  #pragma unroll
  for (int e = 0; e < 4; ++e) { float v = (l[e] >= kth) ? __expf(l[e] - m1) : 0.f; w[e] = v; s += v; }
  float inv = 1.f / s;
  #pragma unroll
  for (int e = 0; e < 4; ++e) w[e] *= inv;
}

union FragB16 { bf16x16 v; uint4 q[2]; };

__device__ __forceinline__ f32x8 wmma_bf16(const FragB16& a, const FragB16& b, f32x8 c) {
  return __builtin_amdgcn_wmma_f32_16x16x32_bf16(false, a.v, false, b.v, (short)0, c, false, false);
}

// =====================================================================
// Kernel W: split fp32 weights into bf16 hi/lo (L2-resident thereafter)
// =====================================================================
__global__ __launch_bounds__(256) void k_cvtw(
    const float* __restrict__ dw, const float* __restrict__ uw,
    unsigned short* __restrict__ dwh, unsigned short* __restrict__ dwl,
    unsigned short* __restrict__ uwh, unsigned short* __restrict__ uwl)
{
  const int n = NEXP * HID * DIM;
  int i = blockIdx.x * 256 + threadIdx.x;
  if (i < n) {
    splitbf(dw[i], dwh[i], dwl[i]);
  } else {
    int j = i - n;
    if (j < n) splitbf(uw[j], uwh[j], uwl[j]);
  }
}

// =====================================================================
// Kernel A: routers (one wave per token, 32x32 values in registers)
// =====================================================================
__global__ __launch_bounds__(256) void k_router(
    const float* __restrict__ x,
    const float* __restrict__ rw1, const float* __restrict__ rb1,
    const float* __restrict__ rw2, const float* __restrict__ rb2,
    float* __restrict__ w2_out, float* __restrict__ part_out)
{
  __shared__ float rs1[NEXP * DIM];
  __shared__ float rs2[NEXP * DIM];
  __shared__ float sW1[8][NEXP];
  const int tid = threadIdx.x;
  for (int i = tid; i < NEXP * DIM; i += 256) { rs1[i] = rw1[i]; rs2[i] = rw2[i]; }
  __syncthreads();

  const int wave = tid >> 5, lane = tid & 31;
  const int tok = blockIdx.x * 8 + wave;
  const float* xp = x + (size_t)tok * DIM;

  float vals[32]; unsigned keys[32];
  #pragma unroll
  for (int j = 0; j < 32; ++j) { vals[j] = xp[j * 32 + lane]; keys[j] = f2key(vals[j]); }

  float l1[NEXP], l2[NEXP];
  #pragma unroll
  for (int e = 0; e < NEXP; ++e) {
    float a = 0.f;
    #pragma unroll
    for (int j = 0; j < 32; ++j) a += vals[j] * rs1[e * DIM + j * 32 + lane];
    l1[e] = wred_f(a) + rb1[e];
  }
  // router2 input: drop largest 512 of 1024 -> keep x < s[512]
  float thr = key2f(rank_select<32>(keys, 512));
  #pragma unroll
  for (int e = 0; e < NEXP; ++e) {
    float a = 0.f;
    #pragma unroll
    for (int j = 0; j < 32; ++j) {
      float v = vals[j];
      a += ((v < thr) ? v : 0.f) * rs2[e * DIM + j * 32 + lane];
    }
    l2[e] = wred_f(a) + rb2[e];
  }
  float w1[NEXP], w2[NEXP];
  top2softmax(l1, w1);
  top2softmax(l2, w2);
  if (lane < NEXP) w2_out[(size_t)tok * NEXP + lane] = w2[lane];
  if (lane == 0) {
    #pragma unroll
    for (int e = 0; e < NEXP; ++e) sW1[wave][e] = w1[e];
  }
  __syncthreads();
  if (tid < NEXP) {
    float s = 0.f;
    #pragma unroll
    for (int w = 0; w < 8; ++w) s += sW1[w][tid];
    part_out[blockIdx.x * NEXP + tid] = s;   // fixed-order partials: deterministic
  }
}

// =====================================================================
// Kernel B: deterministic reduction of w1 partials -> dropout k per expert
// =====================================================================
__global__ __launch_bounds__(256) void k_stats(const float* __restrict__ part,
                                               int* __restrict__ kd)
{
  __shared__ float red[256];
  const int tid = threadIdx.x;
  const float vlist[4] = {0.25f, 0.5f, 0.25f, 0.5f};
  for (int e = 0; e < NEXP; ++e) {
    float s = 0.f;
    for (int i = tid; i < 4096; i += 256) s += part[i * NEXP + e];
    red[tid] = s; __syncthreads();
    for (int o = 128; o > 0; o >>= 1) { if (tid < o) red[tid] += red[tid + o]; __syncthreads(); }
    if (tid == 0) {
      float mean = red[0] / (float)N_TOK;
      float p2 = vlist[e] + 0.1f * mean;
      int k = (int)floorf(p2 * (float)HID);
      kd[e] = (k < 0) ? 0 : ((k > HID) ? HID : k);
    }
    __syncthreads();
  }
}

// =====================================================================
// Kernel C: fused expert pipeline. 32-token tile/block, x staged once in
// LDS (bf16 hi/lo), down WMMA GEMM -> dropout -> up WMMA GEMM, out
// accumulated across experts in VGPRs.
// =====================================================================
#define XS  1032   // padded row stride of x tile (bf16 elems), 16B-aligned
#define DSS 200    // padded row stride of down tile (bf16 elems)
#define SMEM_C (32*XS*2*2 + 32*HID*4 + 32*DSS*2*2 + 32*4*4)   // 182784 B

__global__ __launch_bounds__(256) void k_expert(
    const float* __restrict__ x, const float* __restrict__ db, const float* __restrict__ ub,
    const float* __restrict__ w2g, const int* __restrict__ kdg,
    const unsigned short* __restrict__ dwh, const unsigned short* __restrict__ dwl,
    const unsigned short* __restrict__ uwh, const unsigned short* __restrict__ uwl,
    float* __restrict__ out)
{
  extern __shared__ __align__(16) char smem[];
  unsigned short* xh    = (unsigned short*)smem;              // [32][XS]
  unsigned short* xl    = xh + 32 * XS;                       // [32][XS]
  float*          downf = (float*)(xl + 32 * XS);             // [32][HID]
  unsigned short* dnh   = (unsigned short*)(downf + 32*HID);  // [32][DSS]
  unsigned short* dnl   = dnh + 32 * DSS;                     // [32][DSS]
  float*          w2s   = (float*)(dnl + 32 * DSS);           // [32][4]

  const int tid = threadIdx.x, wave = tid >> 5, lane = tid & 31;
  const int tok0 = blockIdx.x * 32;

  // stage x tile as bf16 hi/lo (single HBM read of x)
  for (int i = tid; i < 32 * 1024; i += 256) {
    int r = i >> 10, c = i & 1023;
    float v = x[(size_t)(tok0 + r) * DIM + c];
    splitbf(v, xh[r * XS + c], xl[r * XS + c]);
  }
  if (tid < 128) w2s[tid] = w2g[(size_t)tok0 * NEXP + tid];
  __syncthreads();

  // WMMA fragment lane decomposition (CDNA5 ISA 7.12.2)
  const int mA  = lane & 15;             // A: row within tile
  const int kbA = (lane >> 4) * 16;      // A: K half (0 or 16)
  const int nB  = lane & 15;             // B: column
  const int koB = (lane >> 4) * 16;      // B: K half (0 or 16)
  const int crow0 = (lane >> 4) * 8;     // C: row base (0 or 8)

  f32x8 accu[16];                        // persistent out accumulators
  #pragma unroll
  for (int t = 0; t < 16; ++t) accu[t] = (f32x8){};

  for (int e = 0; e < NEXP; ++e) {
    const int kd = kdg[e];

    // ---- phase 1: down = x @ dw[e]^T  (bf16x3 split WMMA) ----
    int tmb[3], tnb[3];
    f32x8 acc3[3];
    #pragma unroll
    for (int s = 0; s < 3; ++s) {
      int T = wave + 8 * s;              // 24 tiles = 2 mb x 12 nb
      tmb[s] = T / 12; tnb[s] = T % 12;
      acc3[s] = (f32x8){};
    }
    for (int ks = 0; ks < DIM; ks += 32) {
      #pragma unroll
      for (int s = 0; s < 3; ++s) {
        const unsigned short* pa = xh + (tmb[s] * 16 + mA) * XS + ks + kbA;
        const unsigned short* pl = xl + (tmb[s] * 16 + mA) * XS + ks + kbA;
        FragB16 Ah, Al, Bh, Bl;
        Ah.q[0] = *(const uint4*)pa;        Ah.q[1] = *(const uint4*)(pa + 8);
        Al.q[0] = *(const uint4*)pl;        Al.q[1] = *(const uint4*)(pl + 8);
        const size_t widx = ((size_t)(e * HID + tnb[s] * 16 + nB)) * DIM + ks + koB;
        Bh.q[0] = *(const uint4*)(dwh + widx); Bh.q[1] = *(const uint4*)(dwh + widx + 8);
        Bl.q[0] = *(const uint4*)(dwl + widx); Bl.q[1] = *(const uint4*)(dwl + widx + 8);
        acc3[s] = wmma_bf16(Ah, Bh, acc3[s]);
        acc3[s] = wmma_bf16(Ah, Bl, acc3[s]);
        acc3[s] = wmma_bf16(Al, Bh, acc3[s]);
      }
      if (ks + 32 < DIM)  // hint the K-stream of weights into cache
        __builtin_prefetch(dwh + ((size_t)(e * HID + tnb[0] * 16 + nB)) * DIM + ks + 32 + koB, 0, 0);
    }
    // bias + relu -> LDS f32
    #pragma unroll
    for (int s = 0; s < 3; ++s) {
      int h = tnb[s] * 16 + nB;
      float bias = db[e * HID + h];
      #pragma unroll
      for (int j = 0; j < 8; ++j) {
        int r = tmb[s] * 16 + crow0 + j;
        float v = acc3[s][j] + bias;
        downf[r * HID + h] = v > 0.f ? v : 0.f;
      }
    }
    __syncthreads();

    // ---- phase 2: per-row rank-k dropout + fold (w2 * 0.8) + bf16 split ----
    for (int r = wave; r < 32; r += 8) {
      float dv[6]; unsigned dk6[6];
      #pragma unroll
      for (int j = 0; j < 6; ++j) { dv[j] = downf[r * HID + j * 32 + lane]; dk6[j] = f2key(dv[j]); }
      const float sc = w2s[r * 4 + e] * ADPT_SCALE;
      const bool dodrop = kd > 0;
      float thr = 0.f;
      if (dodrop) {
        int idx = kd - 1; if (idx > HID - 1) idx = HID - 1;
        thr = key2f(rank_select<6>(dk6, idx));
      }
      #pragma unroll
      for (int j = 0; j < 6; ++j) {
        float v = dv[j];
        if (dodrop && !(v > thr)) v = 0.f;
        float g = v * sc;
        splitbf(g, dnh[r * DSS + j * 32 + lane], dnl[r * DSS + j * 32 + lane]);
      }
    }
    __syncthreads();

    // ---- phase 3: out += down_scaled @ uw[e]^T  (accumulate in VGPRs) ----
    for (int ks = 0; ks < HID; ks += 32) {
      FragB16 Ah[2], Al[2];
      #pragma unroll
      for (int mb = 0; mb < 2; ++mb) {
        const unsigned short* pa = dnh + (mb * 16 + mA) * DSS + ks + kbA;
        const unsigned short* pl = dnl + (mb * 16 + mA) * DSS + ks + kbA;
        Ah[mb].q[0] = *(const uint4*)pa; Ah[mb].q[1] = *(const uint4*)(pa + 8);
        Al[mb].q[0] = *(const uint4*)pl; Al[mb].q[1] = *(const uint4*)(pl + 8);
      }
      #pragma unroll
      for (int q = 0; q < 8; ++q) {
        const int nb = wave * 8 + q;     // 64 n-blocks over 8 waves
        const size_t widx = ((size_t)(e * DIM + nb * 16 + nB)) * HID + ks + koB;
        FragB16 Bh, Bl;
        Bh.q[0] = *(const uint4*)(uwh + widx); Bh.q[1] = *(const uint4*)(uwh + widx + 8);
        Bl.q[0] = *(const uint4*)(uwl + widx); Bl.q[1] = *(const uint4*)(uwl + widx + 8);
        #pragma unroll
        for (int mb = 0; mb < 2; ++mb) {
          const int t = mb * 8 + q;
          accu[t] = wmma_bf16(Ah[mb], Bh, accu[t]);
          accu[t] = wmma_bf16(Ah[mb], Bl, accu[t]);
          accu[t] = wmma_bf16(Al[mb], Bh, accu[t]);
        }
      }
    }
    __syncthreads();
  }

  // ---- epilogue: + SCALE * sum_e w2_e * ub[e][d], single HBM write of out ----
  #pragma unroll
  for (int t = 0; t < 16; ++t) {
    const int mb = t >> 3, q = t & 7;
    const int d = (wave * 8 + q) * 16 + nB;
    float ubv[4];
    #pragma unroll
    for (int e = 0; e < NEXP; ++e) ubv[e] = ub[e * DIM + d];
    #pragma unroll
    for (int j = 0; j < 8; ++j) {
      const int r = mb * 16 + crow0 + j;
      float ubsum = 0.f;
      #pragma unroll
      for (int e = 0; e < NEXP; ++e) ubsum += w2s[r * 4 + e] * ubv[e];
      out[(size_t)(tok0 + r) * DIM + d] = accu[t][j] + ADPT_SCALE * ubsum;
    }
  }
}

// =====================================================================
extern "C" void kernel_launch(void* const* d_in, const int* in_sizes, int n_in,
                              void* d_out, int out_size, void* d_ws, size_t ws_size,
                              hipStream_t stream)
{
  (void)in_sizes; (void)n_in; (void)out_size; (void)ws_size;
  const float* x   = (const float*)d_in[0];
  const float* rw1 = (const float*)d_in[1];
  const float* rb1 = (const float*)d_in[2];
  const float* rw2 = (const float*)d_in[3];
  const float* rb2 = (const float*)d_in[4];
  const float* dw  = (const float*)d_in[5];
  const float* db  = (const float*)d_in[6];
  const float* uw  = (const float*)d_in[7];
  const float* ub  = (const float*)d_in[8];
  float* out = (float*)d_out;
  char* ws = (char*)d_ws;

  float* w2g  = (float*)(ws + OFF_W2);
  float* part = (float*)(ws + OFF_PART);
  int*   kdg  = (int*)(ws + OFF_KD);
  unsigned short* dwh = (unsigned short*)(ws + OFF_DWH);
  unsigned short* dwl = (unsigned short*)(ws + OFF_DWL);
  unsigned short* uwh = (unsigned short*)(ws + OFF_UWH);
  unsigned short* uwl = (unsigned short*)(ws + OFF_UWL);

  const int nw = NEXP * HID * DIM;
  k_cvtw<<<(2 * nw + 255) / 256, 256, 0, stream>>>(dw, uw, dwh, dwl, uwh, uwl);
  k_router<<<N_TOK / 8, 256, 0, stream>>>(x, rw1, rb1, rw2, rb2, w2g, part);
  k_stats<<<1, 256, 0, stream>>>(part, kdg);

  hipFuncSetAttribute(reinterpret_cast<const void*>(k_expert),
                      hipFuncAttributeMaxDynamicSharedMemorySize, SMEM_C);
  k_expert<<<N_TOK / 32, 256, SMEM_C, stream>>>(x, db, ub, w2g, kdg,
                                                dwh, dwl, uwh, uwl, out);
}
